// EncoderLayer_83056077571009
// MI455X (gfx1250) — matrix operand order repaired and checked by reading, not compiled, not run on hardware
//
#include <hip/hip_runtime.h>
#include <hip/hip_bf16.h>
#include <math.h>

// ---------------------------------------------------------------------------
// Problem constants (from the reference)
// ---------------------------------------------------------------------------
constexpr int BATCH    = 2;
constexpr int SEQ      = 2048;
constexpr int DMODEL   = 1024;
constexpr int HEADDIM  = 64;
constexpr int DSTATE   = 64;
constexpr int DCONV    = 4;
constexpr int MLP_INNER= 4096;
constexpr int DIN      = 2 * DMODEL;                  // 2048
constexpr int NH       = DIN / HEADDIM;               // 32
constexpr int CONV_DIM = DIN + 2 * DSTATE;            // 2176
constexpr int DPROJ    = 2 * DIN + 2 * DSTATE + NH;   // 4256
constexpr int MTOK     = BATCH * SEQ;                 // 4096 tokens
constexpr float EPS    = 1e-5f;

typedef __attribute__((ext_vector_type(16))) __bf16 v16bf;
typedef __attribute__((ext_vector_type(8)))  float  v8f;

__device__ __forceinline__ float siluf(float x) { return x / (1.f + __expf(-x)); }
__device__ __forceinline__ float softplusf(float x) {
    return (x > 20.f) ? x : log1pf(__expf(x));
}

// Workgroup barrier that only drains the DS counter (LDS ordering) and leaves
// in-flight global loads pending — unlike __syncthreads(), which emits
// s_wait_loadcnt_dscnt 0 and would stall the software pipeline.
__device__ __forceinline__ void wg_sync_lds() {
    asm volatile("s_wait_dscnt 0x0\n\t"
                 "s_barrier_signal -1\n\t"
                 "s_barrier_wait -1" ::: "memory");
}

// ---------------------------------------------------------------------------
// bf16 WMMA GEMM:  C[M,N] = A[M,K] @ B[K,N]  (+ epilogue)
//   MODE 0: C = AB
//   MODE 1: C = AB + resid
//   MODE 2: C = silu(AB + bias)
//   MODE 3: C = AB + bias + resid
// M multiple of 128; K multiple of 32; N tail handled by address clamping
// (junk B columns only feed masked-out output columns).
// Block tile 128x256, BK=32, 256 threads = 8 waves (2x4), wave tile 64x64
// = 4x4 fragments of v_wmma_f32_16x16x32_bf16 (16 WMMA / 8 frag loads).
// Next K tile is kept in registers (software pipeline); with the DS-only
// barrier its global loads stay in flight across the whole WMMA section.
// ---------------------------------------------------------------------------
constexpr int BM  = 128;
constexpr int BN  = 256;
constexpr int BK  = 32;
constexpr int LDT = 36;   // padded LDS stride in bf16 elements (even -> b32 reads OK)

union FragBF { v16bf v; unsigned int u[8]; };

template<int MODE>
__global__ __launch_bounds__(256)
void gemm_bf16_wmma(const float* __restrict__ A, const float* __restrict__ B,
                    float* __restrict__ C, int Ndim, int Kdim,
                    const float* __restrict__ bias, const float* __restrict__ resid)
{
    __shared__ __bf16 As[BM * LDT];   // [m][k]
    __shared__ __bf16 Bs[BN * LDT];   // [n][k] (transposed tile)

    const int tid  = threadIdx.x;
    const int lane = tid & 31;
    const int wave = tid >> 5;
    const int wm   = (wave >> 2) * 64;     // 0 / 64
    const int wn   = (wave &  3) * 64;     // 0 / 64 / 128 / 192
    const int bm0  = blockIdx.y * BM;
    const int bn0  = blockIdx.x * BN;

    const v8f vzero = {0.f,0.f,0.f,0.f,0.f,0.f,0.f,0.f};
    v8f acc[4][4];
#pragma unroll
    for (int i = 0; i < 4; ++i)
#pragma unroll
        for (int j = 0; j < 4; ++j) acc[i][j] = vzero;

    float4 ra[4];   // staged A chunks (next tile)
    float4 rb[8];   // staged B chunks (next tile)

    auto loadA = [&](int k0) {
#pragma unroll
        for (int c = 0; c < 4; ++c) {
            const int idx = tid + c * 256;     // 0..1023
            const int row = idx >> 3;          // 0..127
            const int kc  = (idx & 7) * 4;     // 0..28
            ra[c] = *(const float4*)(A + (size_t)(bm0 + row) * Kdim + k0 + kc);
        }
    };
    auto loadB = [&](int k0) {
#pragma unroll
        for (int c = 0; c < 8; ++c) {
            const int idx = tid + c * 256;     // 0..2047
            const int k   = idx >> 6;          // 0..31
            const int n   = (idx & 63) * 4;    // 0..252
            int gn = bn0 + n;
            if (gn > Ndim - 4) gn = Ndim - 4;  // clamp: junk cols never stored
            rb[c] = *(const float4*)(B + (size_t)(k0 + k) * Ndim + gn);
        }
    };
    auto stageA = [&]() {
#pragma unroll
        for (int c = 0; c < 4; ++c) {
            const int idx = tid + c * 256;
            const int row = idx >> 3;
            const int kc  = (idx & 7) * 4;
            __bf16* dst = &As[row * LDT + kc];
            dst[0] = (__bf16)ra[c].x; dst[1] = (__bf16)ra[c].y;
            dst[2] = (__bf16)ra[c].z; dst[3] = (__bf16)ra[c].w;
        }
    };
    auto stageB = [&]() {
#pragma unroll
        for (int c = 0; c < 8; ++c) {
            const int idx = tid + c * 256;
            const int k   = idx >> 6;
            const int n   = (idx & 63) * 4;
            Bs[(n + 0) * LDT + k] = (__bf16)rb[c].x;
            Bs[(n + 1) * LDT + k] = (__bf16)rb[c].y;
            Bs[(n + 2) * LDT + k] = (__bf16)rb[c].z;
            Bs[(n + 3) * LDT + k] = (__bf16)rb[c].w;
        }
    };

    loadA(0);
    loadB(0);

    for (int k0 = 0; k0 < Kdim; k0 += BK) {
        stageA();
        stageB();

        // Issue next tile's global loads before the barrier; they stay in
        // flight across both barriers and the WMMA section.
        if (k0 + BK < Kdim) { loadA(k0 + BK); loadB(k0 + BK); }

        wg_sync_lds();

        // ---- B fragments: lane n=lane&15, hi-lanes hold K=16..31 ----
        FragBF fb[4];
        {
            const int coln = lane & 15;
            const int kh   = (lane >> 4) << 4;   // 0 or 16
#pragma unroll
            for (int tn = 0; tn < 4; ++tn) {
                const __bf16* src = &Bs[(wn + tn * 16 + coln) * LDT + kh];
#pragma unroll
                for (int i = 0; i < 8; ++i)
                    fb[tn].u[i] = *(const unsigned int*)(src + 2 * i);
            }
        }
        // ---- A fragments: lane m=lane&15, hi-lanes K offset 16 ----
        const int rowm = lane & 15;
        const int ko   = (lane >> 4) << 4;       // 0 or 16
#pragma unroll
        for (int tm = 0; tm < 4; ++tm) {
            FragBF fa;
            const __bf16* src = &As[(wm + tm * 16 + rowm) * LDT];
#pragma unroll
            for (int i = 0; i < 4; ++i)
                fa.u[i]     = *(const unsigned int*)(src + ko + 2 * i);
#pragma unroll
            for (int i = 0; i < 4; ++i)
                fa.u[4 + i] = *(const unsigned int*)(src + ko + 8 + 2 * i);
#pragma unroll
            for (int tn = 0; tn < 4; ++tn)
                acc[tm][tn] = __builtin_amdgcn_wmma_f32_16x16x32_bf16(
                    false, fa.v, false, fb[tn].v, (short)0, acc[tm][tn], false, false);
        }

        wg_sync_lds();
    }

    // ---- epilogue: C/D layout: VGPR j -> M=j (+8 for hi lanes), N=lane&15 ----
    const int lr = (lane >> 4) << 3;
    const int lc = lane & 15;
#pragma unroll
    for (int tm = 0; tm < 4; ++tm) {
#pragma unroll
        for (int tn = 0; tn < 4; ++tn) {
            const int col = bn0 + wn + tn * 16 + lc;
            if (col >= Ndim) continue;
#pragma unroll
            for (int j = 0; j < 8; ++j) {
                const int row = bm0 + wm + tm * 16 + lr + j;
                float v = acc[tm][tn][j];
                if constexpr (MODE == 1) { v += resid[(size_t)row * Ndim + col]; }
                if constexpr (MODE == 2) { v += bias[col]; v = siluf(v); }
                if constexpr (MODE == 3) { v += bias[col] + resid[(size_t)row * Ndim + col]; }
                C[(size_t)row * Ndim + col] = v;
            }
        }
    }
}

// ---------------------------------------------------------------------------
// dt preparation: dt_sp = softplus(dt + dt_bias), dA = exp(dt_sp * (-exp(A_log)))
// ---------------------------------------------------------------------------
__global__ __launch_bounds__(256)
void dtprep_kernel(const float* __restrict__ zxbcdt,
                   const float* __restrict__ dt_bias,
                   const float* __restrict__ A_log,
                   float* __restrict__ dtsp, float* __restrict__ dAv)
{
    const int idx = blockIdx.x * 256 + threadIdx.x;      // over MTOK*NH
    if (idx >= MTOK * NH) return;
    const int h   = idx & (NH - 1);
    const int tok = idx >> 5;
    const float raw = zxbcdt[(size_t)tok * DPROJ + (DIN + CONV_DIM) + h];
    const float sp  = softplusf(raw + dt_bias[h]);
    const float A   = -__expf(A_log[h]);
    dtsp[idx] = sp;
    dAv[idx]  = __expf(sp * A);
}

// ---------------------------------------------------------------------------
// Depthwise causal conv (DCONV=4) + bias + SiLU over the xBC slice of zxbcdt
// ---------------------------------------------------------------------------
__global__ __launch_bounds__(256)
void conv_silu_kernel(const float* __restrict__ zxbcdt,
                      const float* __restrict__ conv_w,
                      const float* __restrict__ conv_b,
                      float* __restrict__ xBC)
{
    const int idx = blockIdx.x * 256 + threadIdx.x;      // over BATCH*SEQ*CONV_DIM
    if (idx >= BATCH * SEQ * CONV_DIM) return;
    const int c = idx % CONV_DIM;
    const int t = (idx / CONV_DIM) % SEQ;
    const int b = idx / (CONV_DIM * SEQ);
    const float* src = zxbcdt + (size_t)b * SEQ * DPROJ + DIN + c;
    float acc = conv_b[c];
#pragma unroll
    for (int j = 0; j < DCONV; ++j) {
        const int tt = t - (DCONV - 1) + j;
        if (tt >= 0) acc += conv_w[c * DCONV + j] * src[(size_t)tt * DPROJ];
    }
    xBC[idx] = siluf(acc);
}

// ---------------------------------------------------------------------------
// Sequential SSM scan. One block per (batch, head), 64 threads.
// Thread p keeps state row h[p, 0..63] in 64 VGPRs; B_t/C_t broadcast via
// ping-pong LDS (one barrier per step).
// ---------------------------------------------------------------------------
__global__ __launch_bounds__(64)
void ssm_scan_kernel(const float* __restrict__ xBC,
                     const float* __restrict__ dtsp,
                     const float* __restrict__ dAv,
                     const float* __restrict__ D_param,
                     float* __restrict__ yssm)
{
    const int bh = blockIdx.x;
    const int b  = bh >> 5;
    const int h  = bh & 31;
    const int p  = threadIdx.x;            // 0..63

    __shared__ float Bsh[2][DSTATE];
    __shared__ float Csh[2][DSTATE];

    float hstate[DSTATE];
#pragma unroll
    for (int n = 0; n < DSTATE; ++n) hstate[n] = 0.f;

    const float Dp = D_param[h];
    const float* xrow = xBC  + (size_t)b * SEQ * CONV_DIM + h * HEADDIM + p;
    const float* Brow = xBC  + (size_t)b * SEQ * CONV_DIM + DIN;
    const float* Crow = Brow + DSTATE;
    const float* dtr  = dtsp + (size_t)b * SEQ * NH + h;
    const float* dAr  = dAv  + (size_t)b * SEQ * NH + h;
    float*       yrow = yssm + (size_t)b * SEQ * DIN + h * HEADDIM + p;

    for (int t = 0; t < SEQ; ++t) {
        const int pp = t & 1;
        Bsh[pp][p] = Brow[(size_t)t * CONV_DIM + p];
        Csh[pp][p] = Crow[(size_t)t * CONV_DIM + p];
        __syncthreads();

        const float dt = dtr[(size_t)t * NH];
        const float dA = dAr[(size_t)t * NH];
        const float xv = xrow[(size_t)t * CONV_DIM];
        const float coef = dt * xv;
        float y = 0.f;
#pragma unroll
        for (int n = 0; n < DSTATE; ++n) {
            hstate[n] = dA * hstate[n] + coef * Bsh[pp][n];
            y += hstate[n] * Csh[pp][n];
        }
        yrow[(size_t)t * DIN] = y + Dp * xv;
    }
}

// ---------------------------------------------------------------------------
// Gated RMSNorm over DIN=2048: y = rmsnorm(y * silu(z)) * norm_w
// One block (256 threads) per token; each thread handles 8 channels.
// ---------------------------------------------------------------------------
__global__ __launch_bounds__(256)
void gated_rmsnorm_kernel(const float* __restrict__ zxbcdt,
                          const float* __restrict__ norm_w,
                          float* __restrict__ yssm)
{
    const int tok = blockIdx.x;
    const int tid = threadIdx.x;
    const float* z = zxbcdt + (size_t)tok * DPROJ;
    float*       y = yssm   + (size_t)tok * DIN;

    float g[8];
    float ss = 0.f;
#pragma unroll
    for (int i = 0; i < 8; ++i) {
        const int c = tid + i * 256;
        const float zv = z[c];
        const float gv = y[c] * siluf(zv);
        g[i] = gv;
        ss += gv * gv;
    }
    __shared__ float red[256];
    red[tid] = ss;
    __syncthreads();
    for (int s = 128; s > 0; s >>= 1) {
        if (tid < s) red[tid] += red[tid + s];
        __syncthreads();
    }
    const float scale = rsqrtf(red[0] * (1.f / DIN) + EPS);
#pragma unroll
    for (int i = 0; i < 8; ++i) {
        const int c = tid + i * 256;
        y[c] = g[i] * scale * norm_w[c];
    }
}

// ---------------------------------------------------------------------------
// Plain RMSNorm over DMODEL=1024: hn = rmsnorm(h) * rms_w
// ---------------------------------------------------------------------------
__global__ __launch_bounds__(256)
void rmsnorm_kernel(const float* __restrict__ hin,
                    const float* __restrict__ w,
                    float* __restrict__ hout)
{
    const int tok = blockIdx.x;
    const int tid = threadIdx.x;
    const float* x = hin  + (size_t)tok * DMODEL;
    float*       o = hout + (size_t)tok * DMODEL;

    float v[4];
    float ss = 0.f;
#pragma unroll
    for (int i = 0; i < 4; ++i) {
        v[i] = x[tid + i * 256];
        ss += v[i] * v[i];
    }
    __shared__ float red[256];
    red[tid] = ss;
    __syncthreads();
    for (int s = 128; s > 0; s >>= 1) {
        if (tid < s) red[tid] += red[tid + s];
        __syncthreads();
    }
    const float scale = rsqrtf(red[0] * (1.f / DMODEL) + EPS);
#pragma unroll
    for (int i = 0; i < 4; ++i) {
        const int c = tid + i * 256;
        o[c] = v[i] * scale * w[c];
    }
}

// ---------------------------------------------------------------------------
// Host-side launch
// ---------------------------------------------------------------------------
extern "C" void kernel_launch(void* const* d_in, const int* in_sizes, int n_in,
                              void* d_out, int out_size, void* d_ws, size_t ws_size,
                              hipStream_t stream)
{
    const float* x        = (const float*)d_in[0];   // (B,S,DMODEL)
    const float* W_in     = (const float*)d_in[1];   // (DMODEL, DPROJ)
    const float* conv_w   = (const float*)d_in[2];   // (CONV_DIM, DCONV)
    const float* conv_b   = (const float*)d_in[3];   // (CONV_DIM,)
    const float* dt_bias  = (const float*)d_in[4];   // (NH,)
    const float* A_log    = (const float*)d_in[5];   // (NH,)
    const float* D_param  = (const float*)d_in[6];   // (NH,)
    const float* norm_w   = (const float*)d_in[7];   // (DIN,)
    const float* W_out    = (const float*)d_in[8];   // (DIN, DMODEL)
    const float* rms_w    = (const float*)d_in[9];   // (DMODEL,)
    const float* mlp_w1   = (const float*)d_in[10];  // (DMODEL, MLP_INNER)
    const float* mlp_b1   = (const float*)d_in[11];  // (MLP_INNER,)
    const float* mlp_w2   = (const float*)d_in[12];  // (MLP_INNER, DMODEL)
    const float* mlp_b2   = (const float*)d_in[13];  // (DMODEL,)
    float* out = (float*)d_out;

    // Workspace layout (floats)
    float* ws = (float*)d_ws;
    float* zxbcdt = ws;                              size_t off = (size_t)MTOK * DPROJ;
    float* xBC    = ws + off;                        off += (size_t)MTOK * CONV_DIM;
    float* yssm   = ws + off;                        off += (size_t)MTOK * DIN;
    float* hbuf   = ws + off;                        off += (size_t)MTOK * DMODEL;
    float* hn     = ws + off;                        off += (size_t)MTOK * DMODEL;
    float* mid    = ws + off;                        off += (size_t)MTOK * MLP_INNER;
    float* dtsp   = ws + off;                        off += (size_t)MTOK * NH;
    float* dAv    = ws + off;

    const dim3 blk(256);
    const int gm = MTOK / BM;   // 32

    // 1) zxbcdt = x @ W_in   [4096 x 1024 x 4256]
    gemm_bf16_wmma<0><<<dim3((DPROJ + BN - 1) / BN, gm), blk, 0, stream>>>(
        x, W_in, zxbcdt, DPROJ, DMODEL, nullptr, nullptr);

    // 2) dt softplus + dA
    dtprep_kernel<<<(MTOK * NH + 255) / 256, blk, 0, stream>>>(
        zxbcdt, dt_bias, A_log, dtsp, dAv);

    // 3) depthwise conv + silu
    conv_silu_kernel<<<(BATCH * SEQ * CONV_DIM + 255) / 256, blk, 0, stream>>>(
        zxbcdt, conv_w, conv_b, xBC);

    // 4) SSM scan
    ssm_scan_kernel<<<BATCH * NH, dim3(64), 0, stream>>>(
        xBC, dtsp, dAv, D_param, yssm);

    // 5) gated RMSNorm (in-place on yssm)
    gated_rmsnorm_kernel<<<MTOK, blk, 0, stream>>>(zxbcdt, norm_w, yssm);

    // 6) h = x + yssm @ W_out   [4096 x 2048 x 1024]
    gemm_bf16_wmma<1><<<dim3(DMODEL / BN, gm), blk, 0, stream>>>(
        yssm, W_out, hbuf, DMODEL, DIN, nullptr, x);

    // 7) hn = rmsnorm(h) * rms_w
    rmsnorm_kernel<<<MTOK, blk, 0, stream>>>(hbuf, rms_w, hn);

    // 8) mid = silu(hn @ mlp_w1 + b1)   [4096 x 1024 x 4096]
    gemm_bf16_wmma<2><<<dim3(MLP_INNER / BN, gm), blk, 0, stream>>>(
        hn, mlp_w1, mid, MLP_INNER, DMODEL, mlp_b1, nullptr);

    // 9) out = h + mid @ mlp_w2 + b2   [4096 x 4096 x 1024]
    gemm_bf16_wmma<3><<<dim3(DMODEL / BN, gm), blk, 0, stream>>>(
        mid, mlp_w2, out, DMODEL, MLP_INNER, mlp_b2, hbuf);
}
